// MultiHeadAttention_78237124264578
// MI455X (gfx1250) — compile-verified, matrix operand order checked by reading, not run on hardware
//
#include <hip/hip_runtime.h>

#define NIN   512
#define DIM   64
#define NHEAD 8
#define BATCH 2
#define SEQ   2048
#define ROWS  (BATCH * SEQ)   // 4096
#define OUTF  (DIM * NHEAD)   // 512
#define NPAIR (NHEAD * BATCH) // 16 contiguous [SEQ,DIM] slabs after the flat reshape

#define SWLD  40              // LDS row stride (elements): 80B -> conflict-free fragment reads

typedef __attribute__((ext_vector_type(16))) __bf16 v16bf;
typedef __attribute__((ext_vector_type(8)))  __bf16 v8bf;
typedef __attribute__((ext_vector_type(4)))  __bf16 v4bf;
typedef __attribute__((ext_vector_type(8)))  float  v8f;
typedef __attribute__((ext_vector_type(4)))  unsigned int v4u;
typedef __attribute__((ext_vector_type(8)))  int v8i;
typedef __attribute__((ext_vector_type(4)))  int v4i;

// ---- WMMA fragment helpers (wave32, 16x16x32 bf16, layouts per ISA 7.12.2) ----
// A (16x32 MxK) / B (32x16 KxN): lo = lane&15 is the row(M)/col(N) index,
// element e holds K = (e>>3)*16 + (lane>>4)*8 + (e&7).

__device__ __forceinline__ v16bf bfrag_kc(const __bf16* __restrict__ p, int ld) {
  const int lane = threadIdx.x & 31;
  const int lo = lane & 15, hi = lane >> 4;
  const __bf16* r = p + (size_t)lo * ld + hi * 8;
  v8bf l0 = *(const v8bf*)(r);       // K = hi*8 + 0..7
  v8bf l1 = *(const v8bf*)(r + 16);  // K = 16 + hi*8 + 0..7
  return __builtin_shufflevector(l0, l1, 0, 1, 2, 3, 4, 5, 6, 7,
                                 8, 9, 10, 11, 12, 13, 14, 15);
}

// same fragment pattern from an LDS tile with row stride SWLD
__device__ __forceinline__ v16bf lds_bfrag(const __bf16* buf) {
  const int lane = threadIdx.x & 31;
  const int lo = lane & 15, hi = lane >> 4;
  const __bf16* r = buf + lo * SWLD + hi * 8;
  v8bf l0 = *(const v8bf*)(r);
  v8bf l1 = *(const v8bf*)(r + 16);
  return __builtin_shufflevector(l0, l1, 0, 1, 2, 3, 4, 5, 6, 7,
                                 8, 9, 10, 11, 12, 13, 14, 15);
}

// bf16 source, K strided by ld: element (k, lo) at p[k*ld + lo]
__device__ __forceinline__ v16bf bfrag_ks(const __bf16* __restrict__ p, int ld) {
  const int lane = threadIdx.x & 31;
  const int lo = lane & 15, hi = lane >> 4;
  const __bf16* c = p + lo + (size_t)(hi * 8) * ld;
  v16bf f;
#pragma unroll
  for (int e = 0; e < 16; ++e) {
    int k = ((e >> 3) << 4) + (e & 7); // hi*8 folded into base
    f[e] = c[(size_t)k * ld];
  }
  return f;
}

// f32 source, K strided (only for the tiny 64x64 M matrices)
__device__ __forceinline__ v16bf ffrag_ks(const float* __restrict__ p, int ld) {
  const int lane = threadIdx.x & 31;
  const int lo = lane & 15, hi = lane >> 4;
  const float* c = p + lo + (size_t)(hi * 8) * ld;
  v16bf f;
#pragma unroll
  for (int e = 0; e < 16; ++e) {
    int k = ((e >> 3) << 4) + (e & 7);
    f[e] = (__bf16)c[(size_t)k * ld];
  }
  return f;
}

__device__ __forceinline__ v8f wmma_bf16(v16bf a, v16bf b, v8f c) {
  return __builtin_amdgcn_wmma_f32_16x16x32_bf16(false, a, false, b, (short)0, c,
                                                 false, false);
}

// ---- TDM: DMA a 64-row x 32-element bf16 W tile into LDS with row padding ----
// D# per ISA 8.3/8.4: 2D tensor, data_size=1 (2B), tensor_dim0=512 (elements),
// tensor_dim0_stride=512, tile_dim0=32, tile_dim1=64, pad: every 16 DWORDs
// (64B = one 32-elem row) insert 4 DWORDs (16B) -> LDS pitch 40 elements = SWLD.
__device__ __forceinline__ void tdm_load_w(const __bf16* gsrc, unsigned lds_addr) {
  unsigned long long ga = (unsigned long long)(uintptr_t)gsrc;
  v4u g0;
  g0.x = 1u;                                   // count=1, is_restore=0, gather=0
  g0.y = lds_addr;                             // LDS byte address
  g0.z = (unsigned)(ga & 0xFFFFFFFFu);         // global_addr[31:0]
  g0.w = (unsigned)((ga >> 32) & 0x01FFFFFFu)  // global_addr[56:32]
         | (2u << 30);                         // type = 2 ("image")
  v8i g1;
  g1[0] = (1 << 16)    // data_size = 1 -> 2 bytes
        | (1 << 20)    // pad_enable
        | (3 << 22)    // pad_interval: 16 DWORDs
        | (3 << 25);   // pad_amount: 4 DWORDs
  g1[1] = (int)(512u << 16);  // tensor_dim0[15:0] = 512
  g1[2] = (int)(64u << 16);   // tensor_dim1[15:0] = 64
  g1[3] = (int)(32u << 16);   // tile_dim0 = 32
  g1[4] = 64;                 // tile_dim1 = 64 (tile_dim2 = 0)
  g1[5] = 512;                // tensor_dim0_stride = 512
  g1[6] = 0;
  g1[7] = 0;
  v4i z4 = {0, 0, 0, 0};
  v8i z8 = {0, 0, 0, 0, 0, 0, 0, 0};
  __builtin_amdgcn_tensor_load_to_lds(g0, g1, z4, z4, z8, 0);
}

// ---- Kernel 0: f32 -> bf16 conversion (x and the three W matrices) ----
__global__ void __launch_bounds__(256) cvt_bf16_kernel(const float* __restrict__ src,
                                                       __bf16* __restrict__ dst,
                                                       int n4) {
  int i = blockIdx.x * 256 + threadIdx.x;
  if (i < n4) {
    float4 v = ((const float4*)src)[i];
    v4bf o;
    o[0] = (__bf16)v.x; o[1] = (__bf16)v.y; o[2] = (__bf16)v.z; o[3] = (__bf16)v.w;
    ((v4bf*)dst)[i] = o;
  }
}

// ---- Kernel 1: Q/K/V projections  C[4096,512] = x @ W^T + bias, stored bf16 ----
// grid(16, 8, 3), block 256 (8 waves). Wave: 32 rows x 64 cols (8 WMMA tiles).
// W tile staged by the Tensor Data Mover into double-buffered LDS (wave 0 issues,
// TENSORcnt + workgroup barrier hand the buffer to all 8 waves).
__global__ void __launch_bounds__(256) qkv_gemm_kernel(
    const __bf16* __restrict__ xb, const __bf16* __restrict__ Wb3,
    const float* __restrict__ bq, const float* __restrict__ bk,
    const float* __restrict__ bv,
    __bf16* __restrict__ Qb, __bf16* __restrict__ Kb, __bf16* __restrict__ Vb) {
  __shared__ __bf16 sW[2][64 * SWLD];

  const __bf16* W = Wb3 + (size_t)blockIdx.z * OUTF * NIN;
  const float* bias; __bf16* C;
  if (blockIdx.z == 0)      { bias = bq; C = Qb; }
  else if (blockIdx.z == 1) { bias = bk; C = Kb; }
  else                      { bias = bv; C = Vb; }

  const int wave = threadIdx.x >> 5;
  const int lane = threadIdx.x & 31;
  const int lo = lane & 15, hi = lane >> 4;
  const int row0 = blockIdx.x * 256 + wave * 32;
  const int col0 = blockIdx.y * 64;

  const __bf16* wtile = W + (size_t)col0 * NIN;  // tile start; step s adds s*32
  const unsigned lds0 = (unsigned)(uintptr_t)&sW[0][0]; // LDS aperture: low 32 bits
  const unsigned lds1 = (unsigned)(uintptr_t)&sW[1][0]; //  = LDS byte offset

  const __bf16* aptr0 = xb + (size_t)row0 * NIN;
  const __bf16* aptr1 = aptr0 + (size_t)16 * NIN;

  v8f acc[2][4] = {};

  // K-step s uses sW[s&1]; 16 steps of 32.
  if (wave == 0) {
    tdm_load_w(wtile, lds0);                       // stage s=0
    __builtin_amdgcn_s_wait_tensorcnt(0);
  }
  v16bf a0A = bfrag_kc(aptr0, NIN);
  v16bf a1A = bfrag_kc(aptr1, NIN);
  v16bf a0B, a1B;
  __syncthreads();

  auto compute = [&](const __bf16* buf, v16bf aa0, v16bf aa1) {
    v16bf bm[4];
#pragma unroll
    for (int t = 0; t < 4; ++t) bm[t] = lds_bfrag(buf + t * 16 * SWLD);
#pragma unroll
    for (int t = 0; t < 4; ++t) {
      acc[0][t] = wmma_bf16(aa0, bm[t], acc[0][t]);
      acc[1][t] = wmma_bf16(aa1, bm[t], acc[1][t]);
    }
  };

  for (int i = 0; i < 7; ++i) {
    const int s0 = 2 * i;
    // step s0: DMA s0+1 into sW[1], prefetch A(s0+1), compute from sW[0]
    if (wave == 0) tdm_load_w(wtile + (s0 + 1) * 32, lds1);
    a0B = bfrag_kc(aptr0 + (s0 + 1) * 32, NIN);
    a1B = bfrag_kc(aptr1 + (s0 + 1) * 32, NIN);
    compute(sW[0], a0A, a1A);
    if (wave == 0) __builtin_amdgcn_s_wait_tensorcnt(0);
    __syncthreads();
    // step s0+1: DMA s0+2 into sW[0], prefetch A(s0+2), compute from sW[1]
    if (wave == 0) tdm_load_w(wtile + (s0 + 2) * 32, lds0);
    a0A = bfrag_kc(aptr0 + (s0 + 2) * 32, NIN);
    a1A = bfrag_kc(aptr1 + (s0 + 2) * 32, NIN);
    compute(sW[1], a0B, a1B);
    if (wave == 0) __builtin_amdgcn_s_wait_tensorcnt(0);
    __syncthreads();
  }
  // step 14 (sW[0] staged, A regs hold kk=448); DMA step 15 into sW[1]
  if (wave == 0) tdm_load_w(wtile + 15 * 32, lds1);
  a0B = bfrag_kc(aptr0 + 15 * 32, NIN);
  a1B = bfrag_kc(aptr1 + 15 * 32, NIN);
  compute(sW[0], a0A, a1A);
  if (wave == 0) __builtin_amdgcn_s_wait_tensorcnt(0);
  __syncthreads();
  compute(sW[1], a0B, a1B);                        // step 15

#pragma unroll
  for (int u = 0; u < 2; ++u) {
#pragma unroll
    for (int t = 0; t < 4; ++t) {
      const float bias_v = bias[col0 + t * 16 + lo];
      __bf16* outp = C + col0 + t * 16 + lo;
#pragma unroll
      for (int r = 0; r < 8; ++r)
        outp[(size_t)(row0 + u * 16 + r + hi * 8) * OUTF] =
            (__bf16)(acc[u][t][r] + bias_v);
    }
  }
}

// ---- Kernel 2a: zero M (16 x 64 x 64 f32) ----
__global__ void zero_m_kernel(float* __restrict__ M) {
  M[(size_t)blockIdx.x * 256 + threadIdx.x] = 0.0f;
}

// ---- Kernel 2b: M_p += scale * K_p^T V_p over a 256-row sequence chunk ----
// grid(16 pairs, 8 chunks), block 128 (4 waves). Wave w: rows dk [w*16, +16).
__global__ void __launch_bounds__(128) ktv_kernel(const __bf16* __restrict__ K,
                                                  const __bf16* __restrict__ V,
                                                  float* __restrict__ M) {
  const int p = blockIdx.x;
  const __bf16* Kp = K + (size_t)p * SEQ * DIM;
  const __bf16* Vp = V + (size_t)p * SEQ * DIM;
  const int wave = threadIdx.x >> 5;
  const int lane = threadIdx.x & 31;
  const int lo = lane & 15, hi = lane >> 4;
  const int n0 = blockIdx.y * 256;

  v8f acc[4] = {};
#pragma unroll
  for (int s = 0; s < 8; ++s) {
    const int n = n0 + s * 32;
    v16bf a = bfrag_ks(Kp + (size_t)n * DIM + wave * 16, DIM);
#pragma unroll
    for (int t = 0; t < 4; ++t) {
      v16bf bm = bfrag_ks(Vp + (size_t)n * DIM + t * 16, DIM);
      acc[t] = wmma_bf16(a, bm, acc[t]);
    }
  }

  const float scale = 0.125f; // DIM^-0.5
  float* Mp = M + (size_t)p * DIM * DIM;
#pragma unroll
  for (int t = 0; t < 4; ++t)
#pragma unroll
    for (int r = 0; r < 8; ++r)
      unsafeAtomicAdd(&Mp[(size_t)(wave * 16 + r + hi * 8) * DIM + t * 16 + lo],
                      acc[t][r] * scale);
}

// ---- Kernel 3: O_p = Q_p @ M_p, scattered to out[b][n][d*NHEAD + h] ----
// grid(16 row-tiles, 16 pairs), block 256 (8 waves).
__global__ void __launch_bounds__(256) attn_out_kernel(const __bf16* __restrict__ Q,
                                                       const float* __restrict__ M,
                                                       float* __restrict__ out) {
  const int p = blockIdx.y;
  const int h = p >> 1, b = p & 1; // p = h*BATCH + b
  const __bf16* Qp = Q + (size_t)p * SEQ * DIM;
  const float* Mp = M + (size_t)p * DIM * DIM;
  const int wave = threadIdx.x >> 5;
  const int lane = threadIdx.x & 31;
  const int lo = lane & 15, hi = lane >> 4;
  const int row0 = blockIdx.x * 128 + wave * 16;

  v8f acc[4] = {};
#pragma unroll
  for (int kk = 0; kk < DIM; kk += 32) {
    v16bf a = bfrag_kc(Qp + (size_t)row0 * DIM + kk, DIM);
#pragma unroll
    for (int t = 0; t < 4; ++t) {
      v16bf bm = ffrag_ks(Mp + (size_t)kk * DIM + t * 16, DIM);
      acc[t] = wmma_bf16(a, bm, acc[t]);
    }
  }
#pragma unroll
  for (int t = 0; t < 4; ++t) {
    const int d = t * 16 + lo;
#pragma unroll
    for (int r = 0; r < 8; ++r) {
      const int n = row0 + r + hi * 8;
      out[((size_t)b * SEQ + n) * OUTF + d * NHEAD + h] = acc[t][r];
    }
  }
}

extern "C" void kernel_launch(void* const* d_in, const int* in_sizes, int n_in,
                              void* d_out, int out_size, void* d_ws, size_t ws_size,
                              hipStream_t stream) {
  (void)in_sizes; (void)n_in; (void)out_size; (void)ws_size;
  const float* x  = (const float*)d_in[0];
  const float* Wq = (const float*)d_in[1];
  const float* bq = (const float*)d_in[2];
  const float* Wk = (const float*)d_in[3];
  const float* bk = (const float*)d_in[4];
  const float* Wv = (const float*)d_in[5];
  const float* bv = (const float*)d_in[6];
  float* out = (float*)d_out;

  // Workspace: M (256 KB f32) | xb (4 MB bf16) | Wb3 (1.5 MB bf16) | Qb/Kb/Vb (12 MB bf16)
  float*  M   = (float*)d_ws;
  __bf16* xb  = (__bf16*)(M + NPAIR * DIM * DIM);
  __bf16* Wb3 = xb + (size_t)ROWS * NIN;
  __bf16* Qb  = Wb3 + (size_t)3 * OUTF * NIN;
  __bf16* Kb  = Qb + (size_t)ROWS * OUTF;
  __bf16* Vb  = Kb + (size_t)ROWS * OUTF;

  cvt_bf16_kernel<<<dim3((ROWS * NIN / 4) / 256), 256, 0, stream>>>(x, xb, ROWS * NIN / 4);
  cvt_bf16_kernel<<<dim3((OUTF * NIN / 4) / 256), 256, 0, stream>>>(Wq, Wb3, OUTF * NIN / 4);
  cvt_bf16_kernel<<<dim3((OUTF * NIN / 4) / 256), 256, 0, stream>>>(Wk, Wb3 + (size_t)OUTF * NIN, OUTF * NIN / 4);
  cvt_bf16_kernel<<<dim3((OUTF * NIN / 4) / 256), 256, 0, stream>>>(Wv, Wb3 + (size_t)2 * OUTF * NIN, OUTF * NIN / 4);
  zero_m_kernel<<<dim3((NPAIR * DIM * DIM) / 256), 256, 0, stream>>>(M);

  qkv_gemm_kernel<<<dim3(ROWS / 256, OUTF / 64, 3), 256, 0, stream>>>(
      xb, Wb3, bq, bk, bv, Qb, Kb, Vb);
  ktv_kernel<<<dim3(NPAIR, SEQ / 256), 128, 0, stream>>>(Kb, Vb, M);
  attn_out_kernel<<<dim3(SEQ / 128, NPAIR), 256, 0, stream>>>(Qb, M, out);
}
